// Head_62766652064018
// MI455X (gfx1250) — matrix operand order repaired and checked by reading, not compiled, not run on hardware
//
#include <hip/hip_runtime.h>
#include <hip/hip_bf16.h>

typedef __bf16 bf16_t;
typedef __attribute__((ext_vector_type(16))) __bf16 bf16x16;
typedef __attribute__((ext_vector_type(8)))  float  floatx8;

union BF16x16 {
    bf16x16 v;
    unsigned int u[8];
    __bf16 h[16];
};

#define ATTN_B 8
#define ATTN_S 2048
#define ATTN_D 1024
#define ATTN_H 64

// ---------------------------------------------------------------------------
// Stage 1: q/k/v projections.  x[B*S, D] (f32) @ W[D, H] (f32) -> bf16.
// Grid: one workgroup per 16-row M-tile; 4 waves, wave w owns H-slice w*16.
// q is pre-scaled by 1/sqrt(H) = 0.125.  v is stored transposed [B, H, S].
// ---------------------------------------------------------------------------
__global__ __launch_bounds__(128) void proj_qkv_kernel(
    const float* __restrict__ x,
    const float* __restrict__ Wq,
    const float* __restrict__ Wk,
    const float* __restrict__ Wv,
    bf16_t* __restrict__ qo,   // [B*S, H] row-major, scaled
    bf16_t* __restrict__ ko,   // [B*S, H] row-major
    bf16_t* __restrict__ vT)   // [B, H, S]
{
    const int D = ATTN_D, H = ATTN_H, S = ATTN_S;
    const int m0   = blockIdx.x * 16;
    const int wave = threadIdx.x >> 5;
    const int lane = threadIdx.x & 31;
    const int n0   = wave * 16;
    const int row  = lane & 15;
    const int half = lane >> 4;

    floatx8 cq = {}; floatx8 ck = {}; floatx8 cv = {};

    for (int kb = 0; kb < D; kb += 32) {
        // A tile (16x32 bf16): lane holds row `row`, K = half*16 + 2vv (+kb).
        BF16x16 a;
        const float* xrow = x + (size_t)(m0 + row) * D + kb;
        #pragma unroll
        for (int vv = 0; vv < 8; ++vv) {
            const int koff = 16 * half + 2 * vv;
            a.h[2 * vv]     = (__bf16)xrow[koff];
            a.h[2 * vv + 1] = (__bf16)xrow[koff + 1];
        }
        // B tiles (32x16 bf16): n per lane, K = half*16 + 2j (+kb).
        const int n = n0 + (lane & 15);
        BF16x16 bq, bk, bv;
        #pragma unroll
        for (int j = 0; j < 8; ++j) {
            const int kk = kb + half * 16 + 2 * j;
            bq.h[2*j]   = (__bf16)Wq[(size_t)kk * H + n];
            bq.h[2*j+1] = (__bf16)Wq[(size_t)(kk + 1) * H + n];
            bk.h[2*j]   = (__bf16)Wk[(size_t)kk * H + n];
            bk.h[2*j+1] = (__bf16)Wk[(size_t)(kk + 1) * H + n];
            bv.h[2*j]   = (__bf16)Wv[(size_t)kk * H + n];
            bv.h[2*j+1] = (__bf16)Wv[(size_t)(kk + 1) * H + n];
        }
        cq = __builtin_amdgcn_wmma_f32_16x16x32_bf16(false, a.v, false, bq.v,
                                                     (short)0, cq, false, false);
        ck = __builtin_amdgcn_wmma_f32_16x16x32_bf16(false, a.v, false, bk.v,
                                                     (short)0, ck, false, false);
        cv = __builtin_amdgcn_wmma_f32_16x16x32_bf16(false, a.v, false, bv.v,
                                                     (short)0, cv, false, false);
    }

    const int   ncol   = n0 + (lane & 15);
    const float qscale = 0.125f;  // 1/sqrt(64)
    #pragma unroll
    for (int r = 0; r < 8; ++r) {
        const int grow = m0 + r + 8 * half;          // global row in [0, B*S)
        qo[(size_t)grow * H + ncol] = (__bf16)(cq[r] * qscale);
        ko[(size_t)grow * H + ncol] = (__bf16)ck[r];
        const int b = grow / S;
        const int s = grow - b * S;
        vT[((size_t)b * H + ncol) * S + s] = (__bf16)cv[r];
    }
}

// ---------------------------------------------------------------------------
// Stage 2: flash attention, one wave (32 threads) per 16-query block.
// Online softmax over causal key blocks of 32 keys.
// ---------------------------------------------------------------------------
__global__ __launch_bounds__(32) void flash_attn_kernel(
    const bf16_t* __restrict__ q,    // [B*S, H] pre-scaled
    const bf16_t* __restrict__ k,    // [B*S, H]
    const bf16_t* __restrict__ vT,   // [B, H, S]
    float* __restrict__ out)         // [B*S, H]
{
    const int S = ATTN_S, H = ATTN_H;
    __shared__ __align__(16) unsigned int pls[16 * 16];  // 16x32 bf16 P tile

    const int bpb  = S / 16;
    const int b    = blockIdx.x / bpb;
    const int q0   = (blockIdx.x % bpb) * 16;
    const int lane = threadIdx.x & 31;
    const int half = lane >> 4;
    const int l15  = lane & 15;

    // Load the 16x64 query tile in WMMA A-layout (two K=32 chunks).
    const unsigned int* qrow =
        (const unsigned int*)(q + ((size_t)(b * S) + q0 + l15) * H);
    BF16x16 qa[2];
    #pragma unroll
    for (int c = 0; c < 2; ++c) {
        #pragma unroll
        for (int vv = 0; vv < 8; ++vv) {
            const int pk = vv + 8 * half;
            qa[c].u[vv] = qrow[c * 16 + pk];
        }
    }

    floatx8 acc[4] = {{}, {}, {}, {}};
    float m_r[8], l_r[8];
    #pragma unroll
    for (int r = 0; r < 8; ++r) { m_r[r] = -3.0e38f; l_r[r] = 0.0f; }

    const unsigned int* ku  = (const unsigned int*)(k + (size_t)b * S * H);
    const unsigned int* vTu = (const unsigned int*)(vT + (size_t)b * H * S);

    for (int kb = 0; kb <= q0; kb += 32) {
        // ---- scores: two 16x16 C tiles (keys kb..+15 and kb+16..+31) ----
        floatx8 s0 = {}, s1 = {};
        #pragma unroll
        for (int c = 0; c < 2; ++c) {
            BF16x16 b0, b1;
            #pragma unroll
            for (int j = 0; j < 8; ++j) {
                const int hp = c * 16 + half * 8 + j;         // packed h pair
                const int k1 = kb + 16 + l15;
                b0.u[j] = ku[(size_t)(kb + l15) * 32 + hp];
                b1.u[j] = ku[(size_t)(k1 < S ? k1 : S - 1) * 32 + hp];
            }
            s0 = __builtin_amdgcn_wmma_f32_16x16x32_bf16(false, qa[c].v, false,
                    b0.v, (short)0, s0, false, false);
            s1 = __builtin_amdgcn_wmma_f32_16x16x32_bf16(false, qa[c].v, false,
                    b1.v, (short)0, s1, false, false);
        }

        // ---- causal mask + online softmax (row = r + 8*half, col = l15) ----
        __bf16* plsb = (__bf16*)pls;
        #pragma unroll
        for (int r = 0; r < 8; ++r) {
            const int qi = q0 + r + 8 * half;
            float v0 = (kb + l15      <= qi) ? s0[r] : -3.0e38f;
            float v1 = (kb + 16 + l15 <= qi) ? s1[r] : -3.0e38f;

            float mx = fmaxf(v0, v1);
            #pragma unroll
            for (int off = 1; off < 16; off <<= 1)
                mx = fmaxf(mx, __shfl_xor(mx, off, 32));
            const float mnew  = fmaxf(m_r[r], mx);
            const float scale = __expf(m_r[r] - mnew);
            m_r[r] = mnew;

            const float p0 = __expf(v0 - mnew);
            const float p1 = __expf(v1 - mnew);
            float rs = p0 + p1;
            #pragma unroll
            for (int off = 1; off < 16; off <<= 1)
                rs += __shfl_xor(rs, off, 32);
            l_r[r] = l_r[r] * scale + rs;

            #pragma unroll
            for (int nt = 0; nt < 4; ++nt) acc[nt][r] *= scale;

            const int prow = r + 8 * half;
            plsb[prow * 32 + l15]      = (__bf16)p0;
            plsb[prow * 32 + 16 + l15] = (__bf16)p1;
        }
        __syncthreads();  // single wave: compiler fence + S_NOP

        // ---- P in A-layout from LDS ----
        BF16x16 pa;
        #pragma unroll
        for (int vv = 0; vv < 8; ++vv) {
            const int pk = vv + 8 * half;
            pa.u[vv] = pls[l15 * 16 + pk];
        }
        __syncthreads();

        // ---- PV: 4 output tiles over H ----
        #pragma unroll
        for (int nt = 0; nt < 4; ++nt) {
            BF16x16 bv;
            const int h = nt * 16 + l15;
            #pragma unroll
            for (int j = 0; j < 8; ++j) {
                int key = kb + half * 16 + 2 * j;
                if (key > S - 2) key = S - 2;  // masked lanes, value irrelevant
                bv.u[j] = vTu[(size_t)h * (S / 2) + (key >> 1)];
            }
            acc[nt] = __builtin_amdgcn_wmma_f32_16x16x32_bf16(false, pa.v, false,
                        bv.v, (short)0, acc[nt], false, false);
        }
    }

    // ---- normalize and store ----
    #pragma unroll
    for (int nt = 0; nt < 4; ++nt) {
        #pragma unroll
        for (int r = 0; r < 8; ++r) {
            const int qi = q0 + r + 8 * half;
            out[((size_t)b * S + qi) * H + nt * 16 + l15] = acc[nt][r] / l_r[r];
        }
    }
}

extern "C" void kernel_launch(void* const* d_in, const int* in_sizes, int n_in,
                              void* d_out, int out_size, void* d_ws, size_t ws_size,
                              hipStream_t stream) {
    // setup_inputs order: x, Wk, Wq, Wv
    const float* x  = (const float*)d_in[0];
    const float* Wk = (const float*)d_in[1];
    const float* Wq = (const float*)d_in[2];
    const float* Wv = (const float*)d_in[3];
    float* out = (float*)d_out;

    const int B = ATTN_B, S = ATTN_S, H = ATTN_H;
    bf16_t* qws  = (bf16_t*)d_ws;
    bf16_t* kws  = qws + (size_t)B * S * H;
    bf16_t* vTws = kws + (size_t)B * S * H;

    proj_qkv_kernel<<<B * S / 16, 128, 0, stream>>>(x, Wq, Wk, Wv, qws, kws, vTws);
    flash_attn_kernel<<<B * S / 16, 32, 0, stream>>>(qws, kws, vTws, out);
}
